// HolographicRuleBinding_7232724926951
// MI455X (gfx1250) — matrix operand order repaired and checked by reading, not compiled, not run on hardware
//
#include <hip/hip_runtime.h>
#include <hip/hip_bf16.h>

// ---------------------------------------------------------------------------
// HolographicRuleBinding on MI455X (gfx1250, wave32, WMMA)
//   bound[t,n] = out_scale[n] * sum_k (mem[t,:] @ W^T)[k] * r_t[(n-k) mod D]
// Two bf16 WMMA GEMMs (circulant formulation, no FFT):
//   K2: filler = memory @ W^T            (16384 x 2048 x 2048)
//   K3: bound  = filler @ Circ(r_rule)   (tokens grouped by rule)
// ---------------------------------------------------------------------------

typedef __attribute__((ext_vector_type(16))) __bf16 v16bf;
typedef __attribute__((ext_vector_type(8)))  float  v8f;

#define D_DIM  2048
#define NRULES 64
#define NTOK   16384           // B*S = 4*4096
#define KP     40              // padded LDS K-stride (bf16 elems); 80B rows keep
                               // b128 fragment loads 16B-aligned, avoid conflicts

// workspace layout (bytes):
//   [0)        r_bf      : 64 x 2048 bf16 normalized roles       (256 KB)
//   [262144)   offsets   : 65 ints                               (pad to 512 B)
//   [262656)   order     : 16384 ints (tokens grouped by rule)   (64 KB)
//   [328192)   filler_bf : 16384 x 2048 bf16                     (64 MB)
#define WS_OFF_RBF     0
#define WS_OFF_OFFSETS 262144
#define WS_OFF_ORDER   262656
#define WS_OFF_FILLER  328192

// ---- feature probes -------------------------------------------------------
#if defined(__has_builtin)
#  if __has_builtin(__builtin_amdgcn_cvt_pk_bf16_f32)
#    define HAVE_CVT_PK_BF16 1
#  endif
#endif

// Async global->LDS via inline asm (bypasses toolchain-divergent builtin
// signatures; see CDNA5 ISA ch.10 / §15.18, GLOBAL_LOAD_ASYNC_TO_LDS_B128).
#if defined(__gfx1250__)
#  define HAVE_ASYNC_LDS 1
static __device__ __forceinline__ void async_ld_b128(const void* g, void* l) {
    // vdst VGPR holds the LDS byte offset (low 32 bits of the generic pointer);
    // vaddr is the 64-bit global address (VGPR pair).
    asm volatile("global_load_async_to_lds_b128 %0, %1, off"
                 :: "v"((unsigned)(unsigned long long)l),
                    "v"((unsigned long long)g)
                 : "memory");
}
static __device__ __forceinline__ void wait_asynccnt0() {
    asm volatile("s_wait_asynccnt 0x0" ::: "memory");
}
#endif

static __device__ __forceinline__ unsigned short f2bf(float x) {
    unsigned u = __builtin_bit_cast(unsigned, x);
    u = (u + 0x7FFFu + ((u >> 16) & 1u)) >> 16;   // round-to-nearest-even
    return (unsigned short)u;
}

#ifdef HAVE_CVT_PK_BF16
static __device__ __forceinline__ unsigned f2bf_pk(float lo, float hi) {
    return __builtin_bit_cast(unsigned, __builtin_amdgcn_cvt_pk_bf16_f32(lo, hi));
}
#else
static __device__ __forceinline__ unsigned f2bf_pk(float lo, float hi) {
    return (unsigned)f2bf(lo) | ((unsigned)f2bf(hi) << 16);
}
#endif

struct Frag { unsigned u[8]; };                    // 16 bf16 = 8 dwords
static __device__ __forceinline__ v16bf as_v16bf(const Frag& f) {
    return __builtin_bit_cast(v16bf, f);
}
static __device__ __forceinline__ v8f v8f_zero() {
    v8f z = {0.f,0.f,0.f,0.f,0.f,0.f,0.f,0.f};
    return z;
}

// ---------------------------------------------------------------------------
// K0: normalize role rows, emit bf16
// ---------------------------------------------------------------------------
__global__ __launch_bounds__(256)
void role_normalize_kernel(const float* __restrict__ role,
                           unsigned short* __restrict__ r_bf) {
    __shared__ float red[256];
    const int r = blockIdx.x;
    const int t = threadIdx.x;
    const float* src = role + (size_t)r * D_DIM;
    float ss = 0.f;
    for (int i = t; i < D_DIM; i += 256) { float v = src[i]; ss += v * v; }
    red[t] = ss; __syncthreads();
    for (int s = 128; s > 0; s >>= 1) { if (t < s) red[t] += red[t + s]; __syncthreads(); }
    const float inv = 1.0f / fmaxf(sqrtf(red[0]), 1e-12f);
    for (int i = t; i < D_DIM; i += 256)
        r_bf[(size_t)r * D_DIM + i] = f2bf(src[i] * inv);
}

// ---------------------------------------------------------------------------
// K1: counting sort of tokens by rule id (single workgroup; 16K elems)
// ---------------------------------------------------------------------------
__global__ __launch_bounds__(256)
void bucket_kernel(const int* __restrict__ rule_ids,
                   int* __restrict__ offsets,     // [65]
                   int* __restrict__ order) {     // [16384]
    __shared__ int hist[NRULES];
    __shared__ int cursor[NRULES];
    const int t = threadIdx.x;
    if (t < NRULES) hist[t] = 0;
    __syncthreads();
    for (int i = t; i < NTOK; i += 256) atomicAdd(&hist[rule_ids[i]], 1);
    __syncthreads();
    if (t == 0) {
        int acc = 0;
        for (int r = 0; r < NRULES; ++r) { cursor[r] = acc; offsets[r] = acc; acc += hist[r]; }
        offsets[NRULES] = acc;
    }
    __syncthreads();
    for (int i = t; i < NTOK; i += 256) {
        int r = rule_ids[i];
        int pos = atomicAdd(&cursor[r], 1);
        order[pos] = i;
    }
}

// ---------------------------------------------------------------------------
// K2: filler = memory @ W^T   (M=16384, N=2048, K=2048), bf16 WMMA, f32 acc
//     WG tile 64(M) x 256(N), 8 waves as 2x4 grid of 32x64 wave tiles,
//     K-step 32: 8 WMMAs per wave per step vs 6 fragment b128 loads.
// ---------------------------------------------------------------------------
__global__ __launch_bounds__(256)
void gemm_filler_kernel(const float* __restrict__ memory,
                        const float* __restrict__ W,
                        unsigned short* __restrict__ filler_bf) {
    __shared__ unsigned short lds_a[64 * KP];     // [m][k] bf16   (5.0 KB)
    __shared__ unsigned short lds_b[256 * KP];    // [n][k] bf16  (20.0 KB)
    const int m0   = blockIdx.x * 64;
    const int n0   = blockIdx.y * 256;
    const int tid  = threadIdx.x;
    const int lane = tid & 31;
    const int w    = tid >> 5;      // wave id 0..7
    const int wm   = w & 1;         // M half   (rows 32*wm..)
    const int wn   = w >> 1;        // N quarter (cols 64*wn..)
    const int mrow = lane & 15;
    const int kh   = lane >> 4;     // lane half

    v8f acc[2][4];
    #pragma unroll
    for (int a = 0; a < 2; ++a)
        #pragma unroll
        for (int b = 0; b < 4; ++b) acc[a][b] = v8f_zero();

    for (int k0 = 0; k0 < D_DIM; k0 += 32) {
        // stage A: 64 rows x 32 f32 -> bf16  (512 float4 slots)
        #pragma unroll
        for (int j = 0; j < 2; ++j) {
            const int slot = tid + 256 * j;
            const int row  = slot >> 3;
            const int cs   = (slot & 7) * 4;
            const float4 v = *(const float4*)(memory + (size_t)(m0 + row) * D_DIM + k0 + cs);
            uint2 p; p.x = f2bf_pk(v.x, v.y); p.y = f2bf_pk(v.z, v.w);
            *(uint2*)&lds_a[row * KP + cs] = p;
        }
        // stage B: W[n][k] slice, 256 rows x 32 f32 -> bf16 (2048 float4 slots)
        #pragma unroll
        for (int j = 0; j < 8; ++j) {
            const int slot = tid + 256 * j;
            const int row  = slot >> 3;
            const int cs   = (slot & 7) * 4;
            const float4 v = *(const float4*)(W + (size_t)(n0 + row) * D_DIM + k0 + cs);
            uint2 p; p.x = f2bf_pk(v.x, v.y); p.y = f2bf_pk(v.z, v.w);
            *(uint2*)&lds_b[row * KP + cs] = p;
        }
        if (k0 + 32 < D_DIM) {  // global_prefetch_b8 for next K tile
            __builtin_prefetch(memory + (size_t)(m0 + (tid >> 3)) * D_DIM + k0 + 32 + (tid & 7) * 4, 0, 1);
            __builtin_prefetch(W + (size_t)(n0 + tid) * D_DIM + k0 + 32, 0, 1);
        }
        __syncthreads();

        // A fragment: 16-bit A 16x32 layout (lane m=lane%16, kh=lane/16;
        //   dword v -> K = (v%4)*2 + kh*8 + (v/4)*16)
        Frag fa[2], fb[4];
        #pragma unroll
        for (int fm = 0; fm < 2; ++fm) {
            const unsigned short* base = &lds_a[(32 * wm + 16 * fm + mrow) * KP];
            #pragma unroll
            for (int v = 0; v < 8; ++v) {
                const int kloc = ((v & 3) * 2) + kh * 8 + ((v >> 2) * 16);
                fa[fm].u[v] = *(const unsigned*)(base + kloc);
            }
        }
        // B fragment: 16-bit B 32x16 layout (lane n=lane%16, kh=lane/16;
        //   dword v -> K = (v%4)*2 + kh*8 + (v/4)*16)
        #pragma unroll
        for (int fn = 0; fn < 4; ++fn) {
            const unsigned short* base = &lds_b[(64 * wn + 16 * fn + mrow) * KP];
            #pragma unroll
            for (int v = 0; v < 8; ++v) {
                const int kloc = ((v & 3) * 2) + kh * 8 + ((v >> 2) * 16);
                fb[fn].u[v] = *(const unsigned*)(base + kloc);
            }
        }
        #pragma unroll
        for (int fm = 0; fm < 2; ++fm)
            #pragma unroll
            for (int fn = 0; fn < 4; ++fn)
                acc[fm][fn] = __builtin_amdgcn_wmma_f32_16x16x32_bf16(
                    false, as_v16bf(fa[fm]), false, as_v16bf(fb[fn]),
                    (short)0, acc[fm][fn], false, false);
        __syncthreads();
    }

    // epilogue: C/D layout (VGPR g: M = g + 8*kh, N = lane%16), store bf16
    #pragma unroll
    for (int fm = 0; fm < 2; ++fm)
        #pragma unroll
        for (int fn = 0; fn < 4; ++fn) {
            const int col = n0 + 64 * wn + 16 * fn + mrow;
            #pragma unroll
            for (int g = 0; g < 8; ++g) {
                const int row = m0 + 32 * wm + 16 * fm + g + 8 * kh;
                filler_bf[(size_t)row * D_DIM + col] = f2bf(acc[fm][fn][g]);
            }
        }
}

// ---------------------------------------------------------------------------
// K3: bound = filler @ Circulant(r_rule), tokens grouped by rule.
//     Grid (rule, n_block); persistent loop over 64-token tiles of the rule.
//     A rows gathered async into LDS (bf16 already); B tile synthesized from
//     r in LDS: B[k][n] = r[(n-k) mod 2048].
// ---------------------------------------------------------------------------
__global__ __launch_bounds__(256)
void gemm_bind_kernel(const unsigned short* __restrict__ filler_bf,
                      const unsigned short* __restrict__ r_bf,
                      const int* __restrict__ offsets,
                      const int* __restrict__ order,
                      const float* __restrict__ out_scale,
                      float* __restrict__ out) {
    __shared__ unsigned short r_lds[D_DIM];      // normalized role, bf16 (4 KB)
    __shared__ unsigned short lds_a[64 * KP];    // gathered filler tile
    __shared__ int tok[64];
    const int rule = blockIdx.x;
    const int n0   = blockIdx.y * 128;
    const int tid  = threadIdx.x;
    const int lane = tid & 31;
    const int w    = tid >> 5;
    const int wm   = w & 1;
    const int wn   = w >> 1;
    const int mrow = lane & 15;
    const int kh   = lane >> 4;

    const int start = offsets[rule];
    const int end   = offsets[rule + 1];

    // stage r into LDS once (async: 2048 bf16 = 256 x b128)
#ifdef HAVE_ASYNC_LDS
    async_ld_b128(r_bf + (size_t)rule * D_DIM + tid * 8, &r_lds[tid * 8]);
    wait_asynccnt0();
#else
    for (int i = tid; i < D_DIM; i += 256) r_lds[i] = r_bf[(size_t)rule * D_DIM + i];
#endif
    __syncthreads();

    const float scale0 = out_scale[n0 + 32 * wn + mrow];
    const float scale1 = out_scale[n0 + 32 * wn + 16 + mrow];
    const unsigned* filler_u = (const unsigned*)filler_bf;
    (void)filler_u;

    for (int t0 = start; t0 < end; t0 += 64) {
        const int cnt = (end - t0 < 64) ? (end - t0) : 64;
        if (tid < 64) tok[tid] = (tid < cnt) ? order[t0 + tid] : 0;
        __syncthreads();

        v8f acc[2][2];
        for (int a = 0; a < 2; ++a) for (int b = 0; b < 2; ++b) acc[a][b] = v8f_zero();

        for (int k0 = 0; k0 < D_DIM; k0 += 32) {
            // gather A: 64 token rows x 32 bf16 (64 B/row = 4 x b128); pad rows zeroed
#ifdef HAVE_ASYNC_LDS
            {
                const int row = tid >> 2;       // 0..63
                const int q   = tid & 3;        // 16-byte quarter
                if (row < cnt) {
                    async_ld_b128(filler_bf + (size_t)tok[row] * D_DIM + k0 + q * 8,
                                  &lds_a[row * KP + q * 8]);
                } else {
                    uint4 z = {0u, 0u, 0u, 0u};
                    *(uint4*)&lds_a[row * KP + q * 8] = z;
                }
            }
            wait_asynccnt0();
#else
            #pragma unroll
            for (int j = 0; j < 4; ++j) {
                const int slot = tid + 256 * j;
                const int row  = slot >> 4;
                const int dw   = slot & 15;
                unsigned val = 0u;
                if (row < cnt)
                    val = filler_u[(size_t)tok[row] * (D_DIM / 2) + (k0 >> 1) + dw];
                *(unsigned*)&lds_a[row * KP + dw * 2] = val;
            }
#endif
            __syncthreads();

            Frag fa[2], fb[2];
            #pragma unroll
            for (int fm = 0; fm < 2; ++fm) {
                const unsigned short* base = &lds_a[(32 * wm + 16 * fm + mrow) * KP];
                #pragma unroll
                for (int v = 0; v < 8; ++v) {
                    const int kloc = ((v & 3) * 2) + kh * 8 + ((v >> 2) * 16);
                    fa[fm].u[v] = *(const unsigned*)(base + kloc);
                }
            }
            // circulant B fragment straight from r in LDS
            #pragma unroll
            for (int fn = 0; fn < 2; ++fn) {
                const int nabs = n0 + 32 * wn + 16 * fn + mrow;
                #pragma unroll
                for (int v = 0; v < 8; ++v) {
                    const int k = k0 + ((v & 3) * 2) + kh * 8 + ((v >> 2) * 16);
                    const unsigned lo = r_lds[(nabs - k) & (D_DIM - 1)];
                    const unsigned hi = r_lds[(nabs - k - 1) & (D_DIM - 1)];
                    fb[fn].u[v] = lo | (hi << 16);
                }
            }
            #pragma unroll
            for (int fm = 0; fm < 2; ++fm)
                #pragma unroll
                for (int fn = 0; fn < 2; ++fn)
                    acc[fm][fn] = __builtin_amdgcn_wmma_f32_16x16x32_bf16(
                        false, as_v16bf(fa[fm]), false, as_v16bf(fb[fn]),
                        (short)0, acc[fm][fn], false, false);
            __syncthreads();
        }

        // epilogue: scatter scaled rows back to token positions
        #pragma unroll
        for (int fm = 0; fm < 2; ++fm)
            #pragma unroll
            for (int fn = 0; fn < 2; ++fn) {
                const int colabs = n0 + 32 * wn + 16 * fn + mrow;
                const float sc = fn ? scale1 : scale0;
                #pragma unroll
                for (int g = 0; g < 8; ++g) {
                    const int rloc = 32 * wm + 16 * fm + g + 8 * kh;
                    if (rloc < cnt)
                        out[(size_t)tok[rloc] * D_DIM + colabs] = acc[fm][fn][g] * sc;
                }
            }
        __syncthreads();   // tok is rewritten next tile
    }
}

// ---------------------------------------------------------------------------
extern "C" void kernel_launch(void* const* d_in, const int* in_sizes, int n_in,
                              void* d_out, int out_size, void* d_ws, size_t ws_size,
                              hipStream_t stream) {
    const float* memory    = (const float*)d_in[0];   // [16384, 2048] f32
    const int*   rule_ids  = (const int*)  d_in[1];   // [16384] i32
    const float* role      = (const float*)d_in[2];   // [64, 2048] f32
    const float* W         = (const float*)d_in[3];   // [2048, 2048] f32
    const float* out_scale = (const float*)d_in[4];   // [2048] f32
    float*       out       = (float*)d_out;           // [16384, 2048] f32

    char* ws = (char*)d_ws;
    unsigned short* r_bf      = (unsigned short*)(ws + WS_OFF_RBF);
    int*            offsets   = (int*)(ws + WS_OFF_OFFSETS);
    int*            order     = (int*)(ws + WS_OFF_ORDER);
    unsigned short* filler_bf = (unsigned short*)(ws + WS_OFF_FILLER);

    role_normalize_kernel<<<NRULES, 256, 0, stream>>>(role, r_bf);
    bucket_kernel<<<1, 256, 0, stream>>>(rule_ids, offsets, order);
    gemm_filler_kernel<<<dim3(NTOK / 64, D_DIM / 256), 256, 0, stream>>>(memory, W, filler_bf);
    gemm_bind_kernel<<<dim3(NRULES, D_DIM / 128), 256, 0, stream>>>(
        filler_bf, r_bf, offsets, order, out_scale, out);
}
